// MoEFFN_10445360464503
// MI455X (gfx1250) — compile-verified, matrix operand order checked by reading, not run on hardware
//
#include <hip/hip_runtime.h>
#include <hip/hip_bf16.h>
#include <math.h>

// ---------------- problem constants (from reference) ----------------
#define E_EXPERTS 8
#define D_DIM     1024
#define DFF_DIM   4096
#define NTOK      4096   // b*s = 4*1024

typedef __bf16 bf16_t;
typedef __attribute__((ext_vector_type(16))) __bf16 v16bf;
typedef __attribute__((ext_vector_type(8)))  __bf16 v8bf;
typedef __attribute__((ext_vector_type(8)))  float  v8f;

static constexpr int BM = 128;        // M tile per workgroup
static constexpr int BN = 128;        // N tile per workgroup
static constexpr int BK = 32;         // K step (= WMMA bf16 K)
static constexpr int LDSS = 40;       // padded LDS row stride in halves (80B, 16B aligned)

// ---------------- CDNA5 async copy helpers ----------------
// GLOBAL_LOAD_ASYNC_TO_LDS_B128 (GV mode): per-lane 64-bit global addr, per-lane
// LDS byte address (low 32 bits of the generic LDS pointer per the aperture rule).
// Tracked by ASYNCcnt; consumed after s_wait_asynccnt + workgroup barrier.
__device__ __forceinline__ void async_ld_b128(const bf16_t* g, bf16_t* l) {
  uint32_t lds_addr = (uint32_t)(uintptr_t)l;
  asm volatile("global_load_async_to_lds_b128 %0, %1, off"
               :: "v"(lds_addr), "v"(g) : "memory");
}
__device__ __forceinline__ void wait_async0() {
  asm volatile("s_wait_asynccnt 0x0" ::: "memory");
}

// ---------------- helper kernels ----------------

__global__ void __launch_bounds__(256) zero_f32(float* __restrict__ p, long n) {
  long i = (long)blockIdx.x * blockDim.x + threadIdx.x;
  long stride = (long)gridDim.x * blockDim.x;
  for (; i < n; i += stride) p[i] = 0.0f;
}

__global__ void __launch_bounds__(256) cvt_f32_bf16(const float* __restrict__ s,
                                                    bf16_t* __restrict__ d, long n) {
  long i = (long)blockIdx.x * blockDim.x + threadIdx.x;
  long stride = (long)gridDim.x * blockDim.x;
  for (; i < n; i += stride) d[i] = (bf16_t)s[i];
}

// src: [E][K][N] f32 row-major  ->  dst: [E][N][K] bf16 (transposed per expert)
__global__ void __launch_bounds__(256) cvt_transpose(const float* __restrict__ src,
                                                     bf16_t* __restrict__ dst,
                                                     int total, int K, int N) {
  int i = blockIdx.x * blockDim.x + threadIdx.x;
  int stride = gridDim.x * blockDim.x;
  int kn = K * N;   // 4M, fits int32
  for (; i < total; i += stride) {
    int e   = i / kn;
    int rem = i - e * kn;
    int k   = rem / N;
    int n   = rem - k * N;
    dst[(size_t)e * kn + (size_t)n * K + k] = (bf16_t)src[i];
  }
}

// one block per token: 8 waves x (dot over D) -> sigmoid -> top2 -> normalized gates
__global__ void __launch_bounds__(256) moe_gates(const float* __restrict__ x,
                                                 const float* __restrict__ cent,
                                                 float* __restrict__ gates) {
  __shared__ float sLog[E_EXPERTS];
  const int tok  = blockIdx.x;
  const int wave = threadIdx.x >> 5;
  const int lane = threadIdx.x & 31;
  const float* u = x + (size_t)tok * D_DIM;
  const float* c = cent + (size_t)wave * D_DIM;
  float acc = 0.0f;
  for (int i = lane; i < D_DIM; i += 32) acc += u[i] * c[i];
  for (int off = 16; off > 0; off >>= 1) acc += __shfl_down(acc, off, 32);
  if (lane == 0) sLog[wave] = acc;
  __syncthreads();
  if (threadIdx.x == 0) {
    float g[E_EXPERTS];
#pragma unroll
    for (int e = 0; e < E_EXPERTS; ++e) g[e] = 1.0f / (1.0f + expf(-sLog[e]));
    int i0 = 0;
#pragma unroll
    for (int e = 1; e < E_EXPERTS; ++e) if (g[e] > g[i0]) i0 = e;
    int i1 = (i0 == 0) ? 1 : 0;
#pragma unroll
    for (int e = 0; e < E_EXPERTS; ++e) if (e != i0 && g[e] > g[i1]) i1 = e;
    float s = g[i0] + g[i1];
#pragma unroll
    for (int e = 0; e < E_EXPERTS; ++e)
      gates[(size_t)tok * E_EXPERTS + e] = (e == i0 || e == i1) ? (g[e] / s) : 0.0f;
  }
}

// deterministic ascending per-expert token lists (wave e scans all tokens, no atomics)
__global__ void __launch_bounds__(256) moe_build_idx(const float* __restrict__ gates,
                                                     int* __restrict__ idx,
                                                     int* __restrict__ cnt) {
  const int e    = threadIdx.x >> 5;
  const int lane = threadIdx.x & 31;
  int count = 0;
  for (int base = 0; base < NTOK; base += 32) {
    int t = base + lane;
    bool sel = gates[(size_t)t * E_EXPERTS + e] > 0.0f;
    unsigned long long m = __ballot(sel);
    int pre = __popcll(m & ((1ull << lane) - 1ull));
    if (sel) idx[e * NTOK + count + pre] = t;
    count += __popcll(m);
  }
  if (lane == 0) cnt[e] = count;
}

// ---------------- WMMA GEMM core ----------------

// Load a 16x32 bf16 fragment (A or B^T) from an LDS panel with row stride LDSS,
// matching the ISA 16-bit operand layout:
//   lanes 0-15 : row = base+lane,    K = {0..7, 16..23}
//   lanes 16-31: row = base+lane-16, K = {8..15, 24..31}
__device__ __forceinline__ v16bf load_frag(const bf16_t* s, int baseRow, int lane) {
  const int r  = baseRow + (lane & 15);
  const int kb = (lane < 16) ? 0 : 8;
  const bf16_t* p = s + r * LDSS + kb;
  v8bf lo = *(const v8bf*)(p);        // K = kb .. kb+7     (ds_load_b128)
  v8bf hi = *(const v8bf*)(p + 16);   // K = kb+16 .. kb+23 (ds_load_b128)
  v16bf f;
#pragma unroll
  for (int i = 0; i < 8; ++i) { f[i] = lo[i]; f[i + 8] = hi[i]; }
  return f;
}

// GEMM1: h[r, col] = gelu( xb[idx[r], :] @ W1^T[col, :] + b1[col] ), gathered rows
__global__ void __launch_bounds__(256) moe_gemm1(
    const bf16_t* __restrict__ xb,    // [NTOK, D]
    const bf16_t* __restrict__ w1t,   // [E, DFF, D]  (transposed)
    const float*  __restrict__ b1,    // [E, DFF]
    const int* __restrict__ idx, const int* __restrict__ cnt,
    bf16_t* __restrict__ h,           // [NTOK, DFF]  (positional gathered rows)
    int e) {
  __shared__ __align__(16) bf16_t sA[2][BM * LDSS];
  __shared__ __align__(16) bf16_t sB[2][BN * LDSS];
  const int rows  = cnt[e];
  const int mBase = blockIdx.y * BM;
  if (mBase >= rows) return;
  const int nBase = blockIdx.x * BN;
  const int tid  = threadIdx.x;
  const int lane = tid & 31;
  const int wave = tid >> 5;
  const int wr = wave & 1;            // 2 wave-rows  (64 rows each)
  const int wc = wave >> 1;           // 4 wave-cols  (32 cols each)
  const int* myIdx = idx + e * NTOK;

  // each thread owns one LDS row half: row = tid>>1, 16 halves at (tid&1)*16
  const int ldRow = tid >> 1;
  const int ldOff = (tid & 1) * 16;
  const int ar  = mBase + ldRow;
  const int arc = (ar < rows) ? ar : (rows - 1);
  const int tok = myIdx[arc];
  const bf16_t* aSrc = xb + (size_t)tok * D_DIM + ldOff;
  const bf16_t* bSrc = w1t + (size_t)e * DFF_DIM * D_DIM
                           + (size_t)(nBase + ldRow) * D_DIM + ldOff;
  const int ldsLoc = ldRow * LDSS + ldOff;

  constexpr int NIT = D_DIM / BK;     // 32
  // stage-0 preload (async copy straight into LDS, ASYNCcnt-tracked)
  async_ld_b128(aSrc,     &sA[0][ldsLoc]);
  async_ld_b128(aSrc + 8, &sA[0][ldsLoc + 8]);
  async_ld_b128(bSrc,     &sB[0][ldsLoc]);
  async_ld_b128(bSrc + 8, &sB[0][ldsLoc + 8]);
  wait_async0();
  __syncthreads();

  v8f acc[4][2] = {};
  for (int i = 0; i < NIT; ++i) {
    const int cur = i & 1;
    if (i + 1 < NIT) {                // prefetch next stage into other buffer
      const int k1 = (i + 1) * BK;
      bf16_t* aD = &sA[1 - cur][ldsLoc];
      bf16_t* bD = &sB[1 - cur][ldsLoc];
      async_ld_b128(aSrc + k1,     aD);
      async_ld_b128(aSrc + k1 + 8, aD + 8);
      async_ld_b128(bSrc + k1,     bD);
      async_ld_b128(bSrc + k1 + 8, bD + 8);
    }
    v16bf af[4], bg[2];
#pragma unroll
    for (int ms = 0; ms < 4; ++ms) af[ms] = load_frag(sA[cur], wr * 64 + ms * 16, lane);
#pragma unroll
    for (int ns = 0; ns < 2; ++ns) bg[ns] = load_frag(sB[cur], wc * 32 + ns * 16, lane);
#pragma unroll
    for (int ms = 0; ms < 4; ++ms)
#pragma unroll
      for (int ns = 0; ns < 2; ++ns)
        acc[ms][ns] = __builtin_amdgcn_wmma_f32_16x16x32_bf16(
            false, af[ms], false, bg[ns], (short)0, acc[ms][ns], false, false);
    wait_async0();                    // next stage landed
    __syncthreads();                  // + everyone done reading cur
  }

#pragma unroll
  for (int ms = 0; ms < 4; ++ms) {
    const int rloc = mBase + wr * 64 + ms * 16 + ((lane < 16) ? 0 : 8);
#pragma unroll
    for (int ns = 0; ns < 2; ++ns) {
      const int col  = nBase + wc * 32 + ns * 16 + (lane & 15);
      const float bias = b1[e * DFF_DIM + col];
#pragma unroll
      for (int j = 0; j < 8; ++j) {
        const int r = rloc + j;
        if (r < rows) {
          float v = acc[ms][ns][j] + bias;
          float g = 0.5f * v * (1.0f + erff(v * 0.70710678118654752f));  // exact GELU
          h[(size_t)r * DFF_DIM + col] = (bf16_t)g;
        }
      }
    }
  }
}

// GEMM2: out[tok, col] += gate * ( h[r, :] @ W2^T[col, :] + b2[col] )
__global__ void __launch_bounds__(256) moe_gemm2(
    const bf16_t* __restrict__ h,     // [NTOK, DFF] positional rows
    const bf16_t* __restrict__ w2t,   // [E, D, DFF] (transposed)
    const float*  __restrict__ b2,    // [E, D]
    const int* __restrict__ idx, const int* __restrict__ cnt,
    const float* __restrict__ gates,  // [NTOK, E]
    float* __restrict__ out,          // [NTOK, D], pre-zeroed
    int e) {
  __shared__ __align__(16) bf16_t sA[2][BM * LDSS];
  __shared__ __align__(16) bf16_t sB[2][BN * LDSS];
  const int rows  = cnt[e];
  const int mBase = blockIdx.y * BM;
  if (mBase >= rows) return;
  const int nBase = blockIdx.x * BN;
  const int tid  = threadIdx.x;
  const int lane = tid & 31;
  const int wave = tid >> 5;
  const int wr = wave & 1;
  const int wc = wave >> 1;
  const int* myIdx = idx + e * NTOK;

  const int ldRow = tid >> 1;
  const int ldOff = (tid & 1) * 16;
  const bf16_t* aSrc = h + (size_t)(mBase + ldRow) * DFF_DIM + ldOff;
  const bf16_t* bSrc = w2t + (size_t)e * DFF_DIM * D_DIM
                           + (size_t)(nBase + ldRow) * DFF_DIM + ldOff;
  const int ldsLoc = ldRow * LDSS + ldOff;

  constexpr int NIT = DFF_DIM / BK;   // 128
  async_ld_b128(aSrc,     &sA[0][ldsLoc]);
  async_ld_b128(aSrc + 8, &sA[0][ldsLoc + 8]);
  async_ld_b128(bSrc,     &sB[0][ldsLoc]);
  async_ld_b128(bSrc + 8, &sB[0][ldsLoc + 8]);
  wait_async0();
  __syncthreads();

  v8f acc[4][2] = {};
  for (int i = 0; i < NIT; ++i) {
    const int cur = i & 1;
    if (i + 1 < NIT) {
      const int k1 = (i + 1) * BK;
      bf16_t* aD = &sA[1 - cur][ldsLoc];
      bf16_t* bD = &sB[1 - cur][ldsLoc];
      async_ld_b128(aSrc + k1,     aD);
      async_ld_b128(aSrc + k1 + 8, aD + 8);
      async_ld_b128(bSrc + k1,     bD);
      async_ld_b128(bSrc + k1 + 8, bD + 8);
    }
    v16bf af[4], bg[2];
#pragma unroll
    for (int ms = 0; ms < 4; ++ms) af[ms] = load_frag(sA[cur], wr * 64 + ms * 16, lane);
#pragma unroll
    for (int ns = 0; ns < 2; ++ns) bg[ns] = load_frag(sB[cur], wc * 32 + ns * 16, lane);
#pragma unroll
    for (int ms = 0; ms < 4; ++ms)
#pragma unroll
      for (int ns = 0; ns < 2; ++ns)
        acc[ms][ns] = __builtin_amdgcn_wmma_f32_16x16x32_bf16(
            false, af[ms], false, bg[ns], (short)0, acc[ms][ns], false, false);
    wait_async0();
    __syncthreads();
  }

#pragma unroll
  for (int ms = 0; ms < 4; ++ms) {
    const int rloc = mBase + wr * 64 + ms * 16 + ((lane < 16) ? 0 : 8);
#pragma unroll
    for (int ns = 0; ns < 2; ++ns) {
      const int col  = nBase + wc * 32 + ns * 16 + (lane & 15);
      const float bias = b2[e * D_DIM + col];
#pragma unroll
      for (int j = 0; j < 8; ++j) {
        const int r = rloc + j;
        if (r < rows) {
          const int token = myIdx[r];
          const float g   = gates[(size_t)token * E_EXPERTS + e];
          float v = acc[ms][ns][j] + bias;
          float* o = out + (size_t)token * D_DIM + col;
          *o += g * v;   // experts run as sequential launches: no races
        }
      }
    }
  }
}

// ---------------- host launcher ----------------

extern "C" void kernel_launch(void* const* d_in, const int* in_sizes, int n_in,
                              void* d_out, int out_size, void* d_ws, size_t ws_size,
                              hipStream_t stream) {
  (void)in_sizes; (void)n_in; (void)ws_size;
  const float* x    = (const float*)d_in[0];
  const float* cent = (const float*)d_in[1];
  const float* W1   = (const float*)d_in[2];
  const float* b1   = (const float*)d_in[3];
  const float* W2   = (const float*)d_in[4];
  const float* b2   = (const float*)d_in[5];
  float* out = (float*)d_out;

  // workspace carve-up (256B aligned), total ~176 MB
  char* ws = (char*)d_ws;
  size_t off = 0;
  auto carve = [&](size_t bytes) -> void* {
    void* p = ws + off;
    off = (off + bytes + 255) & ~(size_t)255;
    return p;
  };
  int*    cnt   = (int*)carve(E_EXPERTS * sizeof(int));
  int*    idx   = (int*)carve((size_t)E_EXPERTS * NTOK * sizeof(int));
  float*  gates = (float*)carve((size_t)NTOK * E_EXPERTS * sizeof(float));
  bf16_t* xb    = (bf16_t*)carve((size_t)NTOK * D_DIM * sizeof(bf16_t));
  bf16_t* w1t   = (bf16_t*)carve((size_t)E_EXPERTS * D_DIM * DFF_DIM * sizeof(bf16_t));
  bf16_t* w2t   = (bf16_t*)carve((size_t)E_EXPERTS * D_DIM * DFF_DIM * sizeof(bf16_t));
  bf16_t* h     = (bf16_t*)carve((size_t)NTOK * DFF_DIM * sizeof(bf16_t));

  // pre-pass: bf16 convert (+ transpose weights so GEMM B-panels are row-major);
  // bf16 weights (134 MB total) then live in the 192 MB L2 across all GEMM passes.
  cvt_f32_bf16 <<<2048, 256, 0, stream>>>(x, xb, (long)NTOK * D_DIM);
  cvt_transpose<<<4096, 256, 0, stream>>>(W1, w1t, E_EXPERTS * D_DIM * DFF_DIM,
                                          D_DIM, DFF_DIM);
  cvt_transpose<<<4096, 256, 0, stream>>>(W2, w2t, E_EXPERTS * DFF_DIM * D_DIM,
                                          DFF_DIM, D_DIM);
  // routing + deterministic gather lists
  moe_gates    <<<NTOK, 256, 0, stream>>>(x, cent, gates);
  moe_build_idx<<<1,    256, 0, stream>>>(gates, idx, cnt);
  // output accumulator init
  zero_f32     <<<2048, 256, 0, stream>>>(out, (long)out_size);

  // per-expert gathered FFN (zero-gated experts skipped => 4x fewer FLOPs, same result)
  const dim3 blk(256);
  const dim3 g1(DFF_DIM / BN, NTOK / BM);  // 32 x 32 (blocks beyond cnt[e] early-exit)
  const dim3 g2(D_DIM  / BN, NTOK / BM);   //  8 x 32
  for (int e = 0; e < E_EXPERTS; ++e) {
    moe_gemm1<<<g1, blk, 0, stream>>>(xb, w1t, b1, idx, cnt, h, e);
    moe_gemm2<<<g2, blk, 0, stream>>>(h, w2t, b2, idx, cnt, gates, out, e);
  }
}
